// E3NN_NequIP_70471823393436
// MI455X (gfx1250) — matrix operand order repaired and checked by reading, not compiled, not run on hardware
//
#include <hip/hip_runtime.h>

typedef __attribute__((ext_vector_type(16))) _Float16 v16h;
typedef __attribute__((ext_vector_type(8)))  float    v8f;
typedef __attribute__((ext_vector_type(4)))  unsigned v4u;
typedef __attribute__((ext_vector_type(8)))  unsigned v8u;

#define NN 20000
#define NE 240000

__device__ __forceinline__ float d_silu(float x){ return x / (1.f + __expf(-x)); }
__device__ __forceinline__ float d_sig(float x){ return 1.f / (1.f + __expf(-x)); }
__device__ __forceinline__ float d_sus(float t){ return t > 0.f ? __expf(-1.f / t) : 0.f; }

__device__ __forceinline__ unsigned pack2h(float a, float b)
{
    union { _Float16 h[2]; unsigned u; } p;
    p.h[0] = (_Float16)a; p.h[1] = (_Float16)b;
    return p.u;
}

#define IS12 0.28867513459481287f   /* 1/sqrt(12) */

// ---------------------------------------------------------------------------
// Tiled WMMA GEMM: C = act(scale * A@B (+ Res))
// A: MxK row-major f32, B: KxN row-major f32, C/Res: MxN f32.
// Block = 128 threads = 4 waves; 32x32 block tile, one 16x16 WMMA tile/wave.
// M must be a multiple of 32 (true for all call sites: 20000, 240000).
// GUARD=false requires K%32==0 && N%32==0 (branch-free, float2 global loads).
// GUARD=true uses clamp+select (unconditional loads, cndmask zeroing).
// A staged row-major [row][k]; B staged TRANSPOSED [col][k] so both fragments
// are contiguous 16B runs -> ds_load_b128.
// ---------------------------------------------------------------------------
template<int ACT, bool HAS_RES, bool GUARD>
__global__ __launch_bounds__(128)
void gemm_wmma(const float* __restrict__ A, const float* __restrict__ B,
               const float* __restrict__ Res, float* __restrict__ C,
               int M, int Nn, int K, float scale)
{
    __shared__ _Float16 As[32][40];   // [m][k]   row base = 80 B (16B aligned)
    __shared__ _Float16 Bs[32][40];   // [n][k]   (transposed stage)

    const int tid  = threadIdx.x;
    const int wave = tid >> 5;
    const int lane = tid & 31;
    const int lg   = lane >> 4;
    const int lr   = lane & 15;
    const int bm   = blockIdx.y * 32;
    const int bn   = blockIdx.x * 32;
    const int wmRow = (wave >> 1) * 16;
    const int wnCol = (wave & 1) * 16;

    v8f acc = {};

    for (int k0 = 0; k0 < K; k0 += 32) {
        // ---- stage A: 512 packed uints, 4 per thread, coalesced ----
        #pragma unroll
        for (int t = 0; t < 4; ++t) {
            int i  = tid + t * 128;        // 0..511
            int r  = i >> 4;               // 0..31  (m row)
            int c2 = (i & 15) << 1;        // 0,2,..,30 (k pair)
            int gm = bm + r;
            int gk = k0 + c2;
            float a0v, a1v;
            if (GUARD) {
                int k0c = gk     < K ? gk     : 0;
                int k1c = gk + 1 < K ? gk + 1 : 0;
                a0v = A[(size_t)gm * K + k0c]; if (gk     >= K) a0v = 0.f;
                a1v = A[(size_t)gm * K + k1c]; if (gk + 1 >= K) a1v = 0.f;
            } else {
                const float2 av = *(const float2*)(A + (size_t)gm * K + gk);
                a0v = av.x; a1v = av.y;
            }
            ((unsigned*)&As[r][0])[c2 >> 1] = pack2h(a0v, a1v);
        }
        // ---- stage B transposed: Bs[n][k], packed along k ----
        #pragma unroll
        for (int t = 0; t < 4; ++t) {
            int i  = tid + t * 128;
            int c  = i & 31;               // n col (coalesced across lanes)
            int r2 = i >> 5;               // 0..15 (k pair)
            int r  = r2 << 1;
            int gn = bn + c;
            int gk = k0 + r;
            float b0v, b1v;
            if (GUARD) {
                int nc  = gn < Nn ? gn : 0;
                int k0c = gk     < K ? gk     : 0;
                int k1c = gk + 1 < K ? gk + 1 : 0;
                b0v = B[(size_t)k0c * Nn + nc]; if (gk     >= K || gn >= Nn) b0v = 0.f;
                b1v = B[(size_t)k1c * Nn + nc]; if (gk + 1 >= K || gn >= Nn) b1v = 0.f;
            } else {
                b0v = B[(size_t)gk       * Nn + gn];
                b1v = B[(size_t)(gk + 1) * Nn + gn];
            }
            ((unsigned*)&Bs[c][0])[r2] = pack2h(b0v, b1v);
        }
        __syncthreads();

        // ---- fragments: 2x b128 LDS loads each (CDNA5 16-bit layouts) ----
        // A 16x32: lane(lr) = row, uint v holds K = {2v,2v+1} + 8*lg (v<4),
        //          {16+2(v-4), ...} + 8*lg (v>=4)  -> uints [4lg..4lg+3], [8+4lg..]
        const unsigned* arow = (const unsigned*)&As[wmRow + lr][0];
        v4u alo = *(const v4u*)(arow + 4 * lg);
        v4u ahi = *(const v4u*)(arow + 8 + 4 * lg);
        v8u au  = __builtin_shufflevector(alo, ahi, 0, 1, 2, 3, 4, 5, 6, 7);
        v16h af = __builtin_bit_cast(v16h, au);
        // B 32x16: lane(lr) = col, same K mapping as A
        //          -> uints [4lg..4lg+3], [8+4lg..]
        const unsigned* brow = (const unsigned*)&Bs[wnCol + lr][0];
        v4u blo = *(const v4u*)(brow + 4 * lg);
        v4u bhi = *(const v4u*)(brow + 8 + 4 * lg);
        v8u bu  = __builtin_shufflevector(blo, bhi, 0, 1, 2, 3, 4, 5, 6, 7);
        v16h bf = __builtin_bit_cast(v16h, bu);

        acc = __builtin_amdgcn_wmma_f32_16x16x32_f16(
                  false, af, false, bf, (short)0, acc, false, false);
        __syncthreads();
    }

    // C/D layout: VGPR v of lane -> (m = v + 8*lg, n = lr)
    const int gn = bn + wnCol + lr;
    #pragma unroll
    for (int v = 0; v < 8; ++v) {
        int gm = bm + wmRow + v + 8 * lg;   // always < M (M % 32 == 0)
        if (!GUARD || gn < Nn) {
            float val = acc[v] * scale;
            if (HAS_RES) val += Res[(size_t)gm * Nn + gn];
            if (ACT == 1) val = d_silu(val);
            C[(size_t)gm * Nn + gn] = val;
        }
    }
}

static void launch_gemm(const float* A, const float* B, const float* Res, float* C,
                        int M, int Nn, int K, float scale, int act, hipStream_t s)
{
    dim3 grid((Nn + 31) / 32, (M + 31) / 32), block(128);
    bool guard = (K % 32) || (Nn % 32);
    if (guard) {
        if (Res)      gemm_wmma<0, true , true ><<<grid, block, 0, s>>>(A, B, Res,     C, M, Nn, K, scale);
        else if (act) gemm_wmma<1, false, true ><<<grid, block, 0, s>>>(A, B, nullptr, C, M, Nn, K, scale);
        else          gemm_wmma<0, false, true ><<<grid, block, 0, s>>>(A, B, nullptr, C, M, Nn, K, scale);
    } else {
        if (Res)      gemm_wmma<0, true , false><<<grid, block, 0, s>>>(A, B, Res,     C, M, Nn, K, scale);
        else if (act) gemm_wmma<1, false, false><<<grid, block, 0, s>>>(A, B, nullptr, C, M, Nn, K, scale);
        else          gemm_wmma<0, false, false><<<grid, block, 0, s>>>(A, B, nullptr, C, M, Nn, K, scale);
    }
}

// ---------------------------------------------------------------------------
// Elementwise / graph kernels
// ---------------------------------------------------------------------------
__global__ void k_zero(float* __restrict__ p, long long n)
{
    long long i = (long long)blockIdx.x * blockDim.x + threadIdx.x;
    if (i < n) p[i] = 0.f;
}

__global__ void k_embed(const int* __restrict__ x, const float* __restrict__ emb,
                        float* __restrict__ h0, int N)
{
    int i = blockIdx.x * blockDim.x + threadIdx.x;
    if (i >= N * 64) return;
    int n = i >> 6, c = i & 63;
    h0[i] = emb[x[n] * 64 + c];
}

// Per-edge: spherical harmonics (l=1,2) and 16-bin radial basis.
__global__ void k_geom(const float* __restrict__ ev, float* __restrict__ sh,
                       float* __restrict__ hb, int E)
{
    int e = blockIdx.x * blockDim.x + threadIdx.x;
    if (e >= E) return;
    float vx = ev[(size_t)e * 3 + 0], vy = ev[(size_t)e * 3 + 1], vz = ev[(size_t)e * 3 + 2];
    float r  = sqrtf(vx * vx + vy * vy + vz * vz);
    float iv = 1.f / r;
    float ux = vx * iv, uy = vy * iv, uz = vz * iv;
    const float S3 = 1.7320508075688772f, S15 = 3.8729833462074170f, S5 = 2.2360679774997896f;
    float* s = sh + (size_t)e * 8;
    s[0] = S3 * ux;  s[1] = S3 * uy;  s[2] = S3 * uz;
    s[3] = S15 * ux * uy;
    s[4] = S15 * uy * uz;
    s[5] = 0.5f * S5 * (3.f * uz * uz - 1.f);
    s[6] = S15 * uz * ux;
    s[7] = 0.5f * S15 * (ux * ux - uy * uy);

    const float step  = 3.15f / 17.f;
    const float istep = 17.f / 3.15f;
    const float amp   = 33.7342923f;   // 1.14136 * e^2 * sqrt(16)
    float* h = hb + (size_t)e * 16;
    #pragma unroll
    for (int i = 0; i < 16; ++i) {
        float vi = (float)(i + 1) * step;
        float dd = (r - vi) * istep;
        h[i] = amp * d_sus(1.f + dd) * d_sus(1.f - dd);
    }
}

// out[n,v,k] = scale * sum_u h[n,u,k] * W[u,v]  (+ Res[n,v,k])
__global__ void k_lin1(const float* __restrict__ h, const float* __restrict__ W,
                       const float* __restrict__ Res, float* __restrict__ out,
                       int N, int U, int V, float scale)
{
    int i = blockIdx.x * blockDim.x + threadIdx.x;
    if (i >= N * V * 3) return;
    int k = i % 3; int t = i / 3; int v = t % V; int n = t / V;
    float s = 0.f;
    for (int u = 0; u < U; ++u)
        s += h[((size_t)n * U + u) * 3 + k] * W[u * V + v];
    s *= scale;
    if (Res) s += Res[((size_t)n * V + v) * 3 + k];
    out[((size_t)n * V + v) * 3 + k] = s;
}

// h0 = silu(g0[:, :128]); h1[u,k] = sigmoid(g0[:,128+u]) * g1[u,k]
__global__ void k_gate(const float* __restrict__ g0, const float* __restrict__ g1,
                       float* __restrict__ h0, float* __restrict__ h1, int N)
{
    int i = blockIdx.x * blockDim.x + threadIdx.x;
    if (i >= N * 176) return;
    int n = i / 176, c = i % 176;
    if (c < 128) {
        float v = g0[(size_t)n * 144 + c];
        h0[(size_t)n * 128 + c] = d_silu(v);
    } else {
        int t = c - 128, u = t / 3, k = t % 3;
        float sg = d_sig(g0[(size_t)n * 144 + 128 + u]);
        h1[((size_t)n * 16 + u) * 3 + k] = sg * g1[((size_t)n * 16 + u) * 3 + k];
    }
}

// ---- Layer 1 messages: u in [0,64) ----
__global__ void k_msg1(const int* __restrict__ src, const int* __restrict__ dst,
                       const float* __restrict__ x0n, const float* __restrict__ w,
                       const float* __restrict__ sh, float* __restrict__ a0,
                       float* __restrict__ a1, int E)
{
    int i = blockIdx.x * blockDim.x + threadIdx.x;
    if (i >= E * 64) return;
    int e = i >> 6, u = i & 63;
    int s = src[e], d = dst[e];
    float x0 = x0n[(size_t)s * 64 + u];
    const float* we = w  + (size_t)e * 128;
    const float* se = sh + (size_t)e * 8;
    atomicAdd(&a0[(size_t)d * 64 + u], x0 * we[u] * IS12);
    float b = x0 * we[64 + u] * IS12;
    atomicAdd(&a1[((size_t)d * 64 + u) * 3 + 0], b * se[0]);
    atomicAdd(&a1[((size_t)d * 64 + u) * 3 + 1], b * se[1]);
    atomicAdd(&a1[((size_t)d * 64 + u) * 3 + 2], b * se[2]);
}

// ---- Layer 2 messages, scalar path: u in [0,128) ----
__global__ void k_msg2a(const int* __restrict__ src, const int* __restrict__ dst,
                        const float* __restrict__ x0n, const float* __restrict__ w,
                        const float* __restrict__ sh, float* __restrict__ a0,
                        float* __restrict__ a1, int E)
{
    int i = blockIdx.x * blockDim.x + threadIdx.x;
    if (i >= E * 128) return;
    int e = i >> 7, u = i & 127;
    int s = src[e], d = dst[e];
    float x0 = x0n[(size_t)s * 128 + u];
    const float* we = w  + (size_t)e * 320;
    const float* se = sh + (size_t)e * 8;
    atomicAdd(&a0[(size_t)d * 144 + u], x0 * we[u] * IS12);
    float b = x0 * we[128 + u] * IS12;
    atomicAdd(&a1[((size_t)d * 176 + u) * 3 + 0], b * se[0]);
    atomicAdd(&a1[((size_t)d * 176 + u) * 3 + 1], b * se[1]);
    atomicAdd(&a1[((size_t)d * 176 + u) * 3 + 2], b * se[2]);
}

// ---- Layer 2 messages, l=1 path: u in [0,16) ----
__global__ void k_msg2b(const int* __restrict__ src, const int* __restrict__ dst,
                        const float* __restrict__ x1n, const float* __restrict__ w,
                        const float* __restrict__ sh, float* __restrict__ a0,
                        float* __restrict__ a1, int E)
{
    int i = blockIdx.x * blockDim.x + threadIdx.x;
    if (i >= E * 16) return;
    int e = i >> 4, u = i & 15;
    int s = src[e], d = dst[e];
    float xx = x1n[((size_t)s * 16 + u) * 3 + 0];
    float xy = x1n[((size_t)s * 16 + u) * 3 + 1];
    float xz = x1n[((size_t)s * 16 + u) * 3 + 2];
    const float* we = w  + (size_t)e * 320;
    const float* se = sh + (size_t)e * 8;
    float sx = se[0], sy = se[1], sz = se[2];

    // m0b : (x1 . sh1)/sqrt(3) * w[272..288)
    float dot = xx * sx + xy * sy + xz * sz;
    atomicAdd(&a0[(size_t)d * 144 + 128 + u], dot * 0.57735026919f * we[272 + u] * IS12);

    // m1b : x1 * w[256..272)
    float wb = we[256 + u] * IS12;
    atomicAdd(&a1[((size_t)d * 176 + 128 + u) * 3 + 0], xx * wb);
    atomicAdd(&a1[((size_t)d * 176 + 128 + u) * 3 + 1], xy * wb);
    atomicAdd(&a1[((size_t)d * 176 + 128 + u) * 3 + 2], xz * wb);

    // m1c : cross(x1, sh1)/sqrt(6) * w[288..304)
    float wc = we[288 + u] * 0.40824829046f * IS12;
    atomicAdd(&a1[((size_t)d * 176 + 144 + u) * 3 + 0], (xy * sz - xz * sy) * wc);
    atomicAdd(&a1[((size_t)d * 176 + 144 + u) * 3 + 1], (xz * sx - xx * sz) * wc);
    atomicAdd(&a1[((size_t)d * 176 + 144 + u) * 3 + 2], (xx * sy - xy * sx) * wc);

    // m1d : CG(1x2->1) contraction with sh2, normalized CG121 hardcoded
    const float C1 = 0.31622776601f;   // sqrt(0.1)
    const float C2 = 0.18257418584f;   // 1/sqrt(30)
    const float C3 = 0.36514837167f;   // sqrt(2/15)
    float q0 = se[3], q1 = se[4], q2 = se[5], q3 = se[6], q4 = se[7];
    float T00 = -C2 * q2 + C1 * q4, T01 = C1 * q0,            T02 = C1 * q3;
    float T10 =  C1 * q0,           T11 = -C2 * q2 - C1 * q4, T12 = C1 * q1;
    float T20 =  C1 * q3,           T21 = C1 * q1,            T22 = C3 * q2;
    float wd = we[304 + u] * IS12;
    atomicAdd(&a1[((size_t)d * 176 + 160 + u) * 3 + 0], (xx * T00 + xy * T10 + xz * T20) * wd);
    atomicAdd(&a1[((size_t)d * 176 + 160 + u) * 3 + 1], (xx * T01 + xy * T11 + xz * T21) * wd);
    atomicAdd(&a1[((size_t)d * 176 + 160 + u) * 3 + 2], (xx * T02 + xy * T12 + xz * T22) * wd);
}

// ---- Output layer messages: u in [0,144) ----
__global__ void k_msgO(const int* __restrict__ src, const int* __restrict__ dst,
                       const float* __restrict__ x0n, const float* __restrict__ x1n,
                       const float* __restrict__ w, const float* __restrict__ sh,
                       float* __restrict__ a0, int E)
{
    int i = blockIdx.x * blockDim.x + threadIdx.x;
    if (i >= E * 144) return;
    int e = i / 144, u = i % 144;
    int s = src[e], d = dst[e];
    const float* we = w  + (size_t)e * 144;
    const float* se = sh + (size_t)e * 8;
    if (u < 128) {
        atomicAdd(&a0[(size_t)d * 144 + u], x0n[(size_t)s * 128 + u] * we[u] * IS12);
    } else {
        int uu = u - 128;
        float dot = x1n[((size_t)s * 16 + uu) * 3 + 0] * se[0]
                  + x1n[((size_t)s * 16 + uu) * 3 + 1] * se[1]
                  + x1n[((size_t)s * 16 + uu) * 3 + 2] * se[2];
        atomicAdd(&a0[(size_t)d * 144 + u], dot * 0.57735026919f * we[u] * IS12);
    }
}

// ---------------------------------------------------------------------------
extern "C" void kernel_launch(void* const* d_in, const int* in_sizes, int n_in,
                              void* d_out, int out_size, void* d_ws, size_t ws_size,
                              hipStream_t stream)
{
    (void)in_sizes; (void)n_in; (void)out_size; (void)ws_size;
    const int N = NN, E = NE;

    const int*   xsp  = (const int*)  d_in[0];
    const int*   eidx = (const int*)  d_in[1];
    const float* evec = (const float*)d_in[2];
    const float* emb  = (const float*)d_in[3];
    const float* L1_lin1_0 = (const float*)d_in[4];
    const float* L1_W1     = (const float*)d_in[5];
    const float* L1_W2     = (const float*)d_in[6];
    const float* L1_lin2_0 = (const float*)d_in[7];
    const float* L1_lin2_1 = (const float*)d_in[8];
    const float* L1_sc_0   = (const float*)d_in[9];
    const float* L2_lin1_0 = (const float*)d_in[10];
    const float* L2_lin1_1 = (const float*)d_in[11];
    const float* L2_W1     = (const float*)d_in[12];
    const float* L2_W2     = (const float*)d_in[13];
    const float* L2_lin2_0 = (const float*)d_in[14];
    const float* L2_lin2_1 = (const float*)d_in[15];
    const float* L2_sc_0   = (const float*)d_in[16];
    const float* L2_sc_1   = (const float*)d_in[17];
    const float* LO_lin1_0 = (const float*)d_in[18];
    const float* LO_lin1_1 = (const float*)d_in[19];
    const float* LO_W1     = (const float*)d_in[20];
    const float* LO_W2     = (const float*)d_in[21];
    const float* LO_lin2_0 = (const float*)d_in[22];
    const float* LO_sc_0   = (const float*)d_in[23];
    const int* src = eidx;
    const int* dst = eidx + E;
    float* out = (float*)d_out;

    // workspace carve-out
    float* ws = (float*)d_ws;
    size_t off = 0;
    auto alloc = [&](size_t n) { float* p = ws + off; off += n; return p; };
    float* sh   = alloc((size_t)E * 8);     // sh1(3) + sh2(5)
    float* hb   = alloc((size_t)E * 16);    // radial basis
    float* rhid = alloc((size_t)E * 64);    // radial-MLP hidden
    float* wed  = alloc((size_t)E * 320);   // radial-MLP output (max width)
    float* x0n  = alloc((size_t)N * 128);   // node scalar features pre-gather
    float* x1n  = alloc((size_t)N * 48);    // node l=1 features pre-gather
    float* h0   = alloc((size_t)N * 128);
    float* h1   = alloc((size_t)N * 48);
    float* sc0  = alloc((size_t)N * 144);
    float* sc1  = alloc((size_t)N * 48);
    float* a0   = alloc((size_t)N * 144);
    float* a1   = alloc((size_t)N * 528);   // 176 x 3
    float* g0   = alloc((size_t)N * 144);
    float* g1   = alloc((size_t)N * 48);

    const float R128 = 0.08838834764831845f;   // 1/sqrt(128)
    const float R176 = 0.07537783614444091f;   // 1/sqrt(176)

    // -------- shared precompute --------
    k_embed<<<(N * 64 + 255) / 256, 256, 0, stream>>>(xsp, emb, h0, N);
    k_geom <<<(E + 255) / 256,     256, 0, stream>>>(evec, sh, hb, E);

    // -------- Layer 1 --------
    launch_gemm(h0, L1_sc_0,   nullptr, sc0,  N, 144, 64, 0.125f, 0, stream);
    launch_gemm(h0, L1_lin1_0, nullptr, x0n,  N,  64, 64, 0.125f, 0, stream);
    launch_gemm(hb, L1_W1,     nullptr, rhid, E,  64, 16, 0.25f,  1, stream);
    launch_gemm(rhid, L1_W2,   nullptr, wed,  E, 128, 64, 0.125f, 0, stream);
    k_zero<<<((long long)N * 64  + 255) / 256, 256, 0, stream>>>(a0, (long long)N * 64);
    k_zero<<<((long long)N * 192 + 255) / 256, 256, 0, stream>>>(a1, (long long)N * 192);
    k_msg1<<<(E * 64 + 255) / 256, 256, 0, stream>>>(src, dst, x0n, wed, sh, a0, a1, E);
    launch_gemm(a0, L1_lin2_0, sc0, g0, N, 144, 64, 0.125f, 0, stream);
    k_lin1<<<(N * 16 * 3 + 255) / 256, 256, 0, stream>>>(a1, L1_lin2_1, nullptr, g1, N, 64, 16, 0.125f);
    k_gate<<<(N * 176 + 255) / 256, 256, 0, stream>>>(g0, g1, h0, h1, N);

    // -------- Layer 2 --------
    launch_gemm(h0, L2_sc_0,   nullptr, sc0,  N, 144, 128, R128, 0, stream);
    k_lin1<<<(N * 16 * 3 + 255) / 256, 256, 0, stream>>>(h1, L2_sc_1, nullptr, sc1, N, 16, 16, 0.25f);
    launch_gemm(h0, L2_lin1_0, nullptr, x0n,  N, 128, 128, R128, 0, stream);
    k_lin1<<<(N * 16 * 3 + 255) / 256, 256, 0, stream>>>(h1, L2_lin1_1, nullptr, x1n, N, 16, 16, 0.25f);
    launch_gemm(hb, L2_W1,     nullptr, rhid, E,  64,  16, 0.25f,  1, stream);
    launch_gemm(rhid, L2_W2,   nullptr, wed,  E, 320,  64, 0.125f, 0, stream);
    k_zero<<<((long long)N * 144 + 255) / 256, 256, 0, stream>>>(a0, (long long)N * 144);
    k_zero<<<((long long)N * 528 + 255) / 256, 256, 0, stream>>>(a1, (long long)N * 528);
    k_msg2a<<<(E * 128 + 255) / 256, 256, 0, stream>>>(src, dst, x0n, wed, sh, a0, a1, E);
    k_msg2b<<<(E * 16  + 255) / 256, 256, 0, stream>>>(src, dst, x1n, wed, sh, a0, a1, E);
    launch_gemm(a0, L2_lin2_0, sc0, g0, N, 144, 144, 1.f / 12.f, 0, stream);
    k_lin1<<<(N * 16 * 3 + 255) / 256, 256, 0, stream>>>(a1, L2_lin2_1, sc1, g1, N, 176, 16, R176);
    k_gate<<<(N * 176 + 255) / 256, 256, 0, stream>>>(g0, g1, h0, h1, N);

    // -------- Output layer --------
    launch_gemm(h0, LO_sc_0,   nullptr, sc0,  N,   4, 128, R128, 0, stream);
    launch_gemm(h0, LO_lin1_0, nullptr, x0n,  N, 128, 128, R128, 0, stream);
    k_lin1<<<(N * 16 * 3 + 255) / 256, 256, 0, stream>>>(h1, LO_lin1_1, nullptr, x1n, N, 16, 16, 0.25f);
    launch_gemm(hb, LO_W1,     nullptr, rhid, E,  64,  16, 0.25f,  1, stream);
    launch_gemm(rhid, LO_W2,   nullptr, wed,  E, 144,  64, 0.125f, 0, stream);
    k_zero<<<((long long)N * 144 + 255) / 256, 256, 0, stream>>>(a0, (long long)N * 144);
    k_msgO<<<(E * 144 + 255) / 256, 256, 0, stream>>>(src, dst, x0n, x1n, wed, sh, a0, E);
    launch_gemm(a0, LO_lin2_0, sc0, out, N, 4, 144, 1.f / 12.f, 0, stream);
}
